// CausalMultiHeadSelfAttention_50414326121010
// MI455X (gfx1250) — matrix operand order repaired and checked by reading, not compiled, not run on hardware
//
#include <hip/hip_runtime.h>

// ---------------------------------------------------------------------------
// Causal MHA with RoPE for MI455X (gfx1250, wave32, WMMA).
// All matmuls via v_wmma_f32_16x16x32_bf16 (bf16 in, f32 accumulate).
// Round 2: 64x64 register-blocked GEMM (16 WMMA / K-step, 4x frag reuse),
//          32-query flash-attention waves (2x K/V fragment reuse).
// ---------------------------------------------------------------------------

typedef __bf16 bf16_t;
typedef __attribute__((ext_vector_type(16))) __bf16 v16bf;
typedef __attribute__((ext_vector_type(8)))  __bf16 v8bf;
typedef __attribute__((ext_vector_type(8)))  float  v8f;

#define D_MODEL 1024
#define N_HEADS 16
#define D_K     64
#define SEQ     2048
#define BATCH   2

union AFrag { v16bf v; v8bf h[2]; };

static __device__ __forceinline__ v8f wmma_bf16(v16bf a, v16bf b, v8f c) {
  // D = A(16x32) * B(32x16) + C, f32 accumulate
  return __builtin_amdgcn_wmma_f32_16x16x32_bf16(
      /*neg_a=*/false, a, /*neg_b=*/false, b,
      /*c_mod=*/(short)0, c, /*reuse_a=*/false, /*reuse_b=*/false);
}

// ---------------------------------------------------------------------------
// fp32 -> bf16 convert
// ---------------------------------------------------------------------------
__global__ void cvt_f32_bf16(const float* __restrict__ in,
                             bf16_t* __restrict__ out, size_t n) {
  size_t i = (size_t)blockIdx.x * blockDim.x + threadIdx.x;
  if (i < n) out[i] = (bf16_t)in[i];
}

// ---------------------------------------------------------------------------
// GEMM (NT): C[M,N] = A[M,K] * B[N,K]^T.  A,B bf16 row-major; C f32 or bf16.
// 64x64 output tile per wave: 4 A-frags x 4 B-frags -> 16 WMMAs per K=32 step
// (each fragment reused 4x; ~32 FLOP/byte from the L2-resident operands).
// ---------------------------------------------------------------------------
template <bool BF16_OUT>
__global__ __launch_bounds__(256) void gemm64_nt(
    const bf16_t* __restrict__ A, const bf16_t* __restrict__ B,
    void* __restrict__ Cout, int M, int N, int K) {
  const int wave = threadIdx.x >> 5;
  const int lane = threadIdx.x & 31;
  const int tile = blockIdx.x * 8 + wave;
  const int tilesN = N >> 6;            // 64-wide column blocks
  const int tm = tile / tilesN;
  const int tn = tile % tilesN;
  if (tm >= (M >> 6)) return;

  const int m    = lane & 15;   // A row within 16-tile / B column within tile
  const int half = lane >> 4;

  // A fragment rows: lane (half*16+m) holds k = half*16 + (0..15) contiguous
  const bf16_t* pa[4];
  // B fragment cols: lane holds column n = m, k = half*16 + (0..15) contiguous
  const bf16_t* pb[4];
#pragma unroll
  for (int i = 0; i < 4; ++i) {
    pa[i] = A + (size_t)(tm * 64 + i * 16 + m) * K + half * 16;
    pb[i] = B + (size_t)(tn * 64 + i * 16 + m) * K + half * 16;
  }

  v8f acc[4][4] = {};
  for (int kb = 0; kb < K; kb += 32) {
    __builtin_prefetch(pa[0] + kb + 512, 0, 0);
    __builtin_prefetch(pb[0] + kb + 512, 0, 0);
    AFrag af[4];
#pragma unroll
    for (int i = 0; i < 4; ++i) {
      af[i].h[0] = *(const v8bf*)(pa[i] + kb);
      af[i].h[1] = *(const v8bf*)(pa[i] + kb + 8);
    }
#pragma unroll
    for (int j = 0; j < 4; ++j) {
      v16bf bv = *(const v16bf*)(pb[j] + kb);
#pragma unroll
      for (int i = 0; i < 4; ++i)
        acc[i][j] = wmma_bf16(af[i].v, bv, acc[i][j]);
    }
  }

  // D layout: VGPR v -> row (half*8 + v), lane%16 -> column
#pragma unroll
  for (int i = 0; i < 4; ++i) {
    const int row0 = tm * 64 + i * 16 + half * 8;
#pragma unroll
    for (int j = 0; j < 4; ++j) {
      const int col = tn * 64 + j * 16 + m;
      if (BF16_OUT) {
        bf16_t* C = (bf16_t*)Cout;
#pragma unroll
        for (int v = 0; v < 8; ++v)
          C[(size_t)(row0 + v) * N + col] = (bf16_t)acc[i][j][v];
      } else {
        float* C = (float*)Cout;
#pragma unroll
        for (int v = 0; v < 8; ++v)
          C[(size_t)(row0 + v) * N + col] = acc[i][j][v];
      }
    }
  }
}

// ---------------------------------------------------------------------------
// RoPE + relayout.
//   do_rope=1: Y[b,s,h*64+d] -> rotate pairs -> out[b,h,s,d]      (Q, K)
//   do_rope=0: Y[b,s,h*64+d] -> out[b,h,d,s]  (transposed V for P*V B-frags)
// One thread per (even,odd) pair.
// ---------------------------------------------------------------------------
__global__ void rope_relayout(const bf16_t* __restrict__ Y,
                              bf16_t* __restrict__ out,
                              const int* __restrict__ pos,
                              int S, int do_rope) {
  size_t idx = (size_t)blockIdx.x * blockDim.x + threadIdx.x;
  const size_t total = (size_t)BATCH * S * N_HEADS * (D_K / 2);
  if (idx >= total) return;
  const int i = (int)(idx & 31);                 // pair index, d = 2i, 2i+1
  const int h = (int)((idx >> 5) & (N_HEADS - 1));
  const size_t bs = idx >> 9;                    // b*S + s
  const int s = (int)(bs % S);
  const int b = (int)(bs / S);

  const bf16_t* p = Y + ((size_t)(b * S + s)) * D_MODEL + h * D_K + 2 * i;
  float x1 = (float)p[0];
  float x2 = (float)p[1];

  if (do_rope) {
    float pf  = (float)pos[s];
    float inv = __powf(10000.0f, -(float)(2 * i) / (float)D_K);
    float ang = pf * inv;
    float sn, cs;
    __sincosf(ang, &sn, &cs);
    float r1 = x1 * cs - x2 * sn;
    float r2 = x1 * sn + x2 * cs;
    bf16_t* q = out + ((size_t)(b * N_HEADS + h) * S + s) * D_K + 2 * i;
    q[0] = (bf16_t)r1;
    q[1] = (bf16_t)r2;
  } else {
    bf16_t* q = out + ((size_t)(b * N_HEADS + h) * D_K + 2 * i) * S + s;
    q[0] = (bf16_t)x1;   // d = 2i   row
    q[S] = (bf16_t)x2;   // d = 2i+1 row
  }
}

// ---------------------------------------------------------------------------
// Flash attention: one wave per (b, h, 32-query tile pair).
//   Qr [B,H,S,64] bf16, Kr [B,H,S,64] bf16, Vt [B,H,64,S] bf16 (transposed)
//   O  [B*S, H*64] bf16   (flat rows for the final projection GEMM)
// Per 32-key block: 8 WMMAs (logits) + 8 WMMAs (P*V); every K/V fragment
// feeds two WMMAs; all 32 lanes own a softmax row.
// ---------------------------------------------------------------------------
__global__ __launch_bounds__(32) void flash_attn(
    const bf16_t* __restrict__ Qr, const bf16_t* __restrict__ Kr,
    const bf16_t* __restrict__ Vt, bf16_t* __restrict__ O, int S) {
  __shared__ float sS[32][32];                    // f32 logits tile (32q x 32k)
  __shared__ __align__(16) bf16_t sP[32][32];     // bf16 probabilities
  __shared__ float sStat[32];                     // alpha / row-sum broadcast

  const int lane = threadIdx.x;
  const int m    = lane & 15;
  const int half = lane >> 4;
  const int b  = blockIdx.z;
  const int h  = blockIdx.y;
  const int q0 = blockIdx.x * 32;

  const bf16_t* Qbh = Qr + (size_t)(b * N_HEADS + h) * S * D_K;
  const bf16_t* Kbh = Kr + (size_t)(b * N_HEADS + h) * S * D_K;
  const bf16_t* Vbh = Vt + (size_t)(b * N_HEADS + h) * D_K * S;

  // Q fragments: 2 query sub-tiles x 2 K-chunks, resident all kernel.
  AFrag qf[2][2];
#pragma unroll
  for (int qt = 0; qt < 2; ++qt)
#pragma unroll
    for (int c = 0; c < 2; ++c) {
      const bf16_t* p =
          Qbh + (size_t)(q0 + qt * 16 + m) * D_K + c * 32 + half * 16;
      qf[qt][c].h[0] = *(const v8bf*)p;
      qf[qt][c].h[1] = *(const v8bf*)(p + 8);
    }

  v8f o[2][4] = {};                 // 32x64 f32 accumulator (2x4 D-tiles)
  float row_m = -__builtin_inff();  // lane owns query row q0+lane
  float row_l = 0.0f;
  const float scale = 0.125f;       // 1/sqrt(64)

  for (int k0 = 0; k0 <= q0 + 31; k0 += 32) {
    // ---- logits: S_tile(32q x 32k) = Q * K^T * scale ----
#pragma unroll
    for (int t = 0; t < 2; ++t) {
      v16bf bv0, bv1;
      {
        const bf16_t* p =
            Kbh + (size_t)(k0 + t * 16 + m) * D_K + half * 16;
        bv0 = *(const v16bf*)p;           // k-chunk 0 (d = 0..31)
        bv1 = *(const v16bf*)(p + 32);    // k-chunk 1 (d = 32..63)
      }
#pragma unroll
      for (int qt = 0; qt < 2; ++qt) {
        v8f acc = {};
        acc = wmma_bf16(qf[qt][0].v, bv0, acc);
        acc = wmma_bf16(qf[qt][1].v, bv1, acc);
#pragma unroll
        for (int v = 0; v < 8; ++v)
          sS[qt * 16 + half * 8 + v][t * 16 + m] = acc[v] * scale;
      }
    }
    __syncthreads();

    // ---- online softmax: lane r owns query row q0+r (all 32 lanes) ----
    {
      const int qrow = q0 + lane;
      float sv[32];
      float mx = row_m;
#pragma unroll
      for (int n = 0; n < 32; ++n) {
        float s = sS[lane][n];
        if (k0 + n > qrow) s = -__builtin_inff();
        sv[n] = s;
        mx = fmaxf(mx, s);
      }
      float alpha = (mx == -__builtin_inff()) ? 0.0f : __expf(row_m - mx);
      float lsum = 0.0f;
#pragma unroll
      for (int n = 0; n < 32; ++n) {
        float pv = (sv[n] == -__builtin_inff()) ? 0.0f : __expf(sv[n] - mx);
        sP[lane][n] = (bf16_t)pv;
        lsum += pv;
      }
      row_l = row_l * alpha + lsum;
      row_m = mx;
      sStat[lane] = alpha;
    }
    __syncthreads();

    // ---- rescale accumulators by alpha(row) ----
#pragma unroll
    for (int qt = 0; qt < 2; ++qt) {
      float al[8];
#pragma unroll
      for (int v = 0; v < 8; ++v) al[v] = sStat[qt * 16 + half * 8 + v];
#pragma unroll
      for (int t = 0; t < 4; ++t)
#pragma unroll
        for (int v = 0; v < 8; ++v) o[qt][t][v] *= al[v];
    }

    // ---- P fragments from LDS (A layout: 2x 16B contiguous per lane) ----
    AFrag pf[2];
#pragma unroll
    for (int qt = 0; qt < 2; ++qt) {
      pf[qt].h[0] = *(const v8bf*)(&sP[qt * 16 + m][half * 16]);
      pf[qt].h[1] = *(const v8bf*)(&sP[qt * 16 + m][half * 16 + 8]);
    }

    // ---- O += P(32x32) * V(32x64): V^T rows contiguous, each frag 2 WMMAs --
#pragma unroll
    for (int t = 0; t < 4; ++t) {
      const bf16_t* p = Vbh + (size_t)(t * 16 + m) * S + k0 + half * 16;
      v16bf bv = *(const v16bf*)p;
#pragma unroll
      for (int qt = 0; qt < 2; ++qt)
        o[qt][t] = wmma_bf16(pf[qt].v, bv, o[qt][t]);
    }
    __syncthreads();
  }

  // ---- epilogue: divide by row sums, store bf16 [B*S, H*64] ----
  sStat[lane] = row_l;
  __syncthreads();
#pragma unroll
  for (int qt = 0; qt < 2; ++qt) {
    float inv[8];
#pragma unroll
    for (int v = 0; v < 8; ++v) {
      float l = sStat[qt * 16 + half * 8 + v];
      inv[v] = (l > 0.0f) ? (1.0f / l) : 0.0f;
    }
#pragma unroll
    for (int t = 0; t < 4; ++t)
#pragma unroll
      for (int v = 0; v < 8; ++v) {
        const int row = q0 + qt * 16 + half * 8 + v;
        const int col = h * D_K + t * 16 + m;
        O[(size_t)(b * S + row) * D_MODEL + col] =
            (bf16_t)(o[qt][t][v] * inv[v]);
      }
  }
}

// ---------------------------------------------------------------------------
// Host-side orchestration (graph-capture safe: async launches only).
// ---------------------------------------------------------------------------
extern "C" void kernel_launch(void* const* d_in, const int* in_sizes, int n_in,
                              void* d_out, int out_size, void* d_ws,
                              size_t ws_size, hipStream_t stream) {
  (void)in_sizes; (void)n_in; (void)out_size; (void)ws_size;
  const float* x  = (const float*)d_in[0];
  const float* Wq = (const float*)d_in[1];
  const float* Wk = (const float*)d_in[2];
  const float* Wv = (const float*)d_in[3];
  const float* Wo = (const float*)d_in[4];
  const int* tpos = (const int*)d_in[5];

  const int S = SEQ, D = D_MODEL;
  const size_t rows = (size_t)BATCH * S;   // 4096
  const size_t matE = rows * D;            // x / Q / K / V / O elements
  const size_t wE   = (size_t)D * D;       // weight elements

  char* p = (char*)d_ws;
  bf16_t* Xb = (bf16_t*)p; p += matE * sizeof(bf16_t);
  bf16_t* Wb = (bf16_t*)p; p += 4 * wE * sizeof(bf16_t);  // Wq,Wk,Wv,Wo bf16
  bf16_t* Qb = (bf16_t*)p; p += matE * sizeof(bf16_t);
  bf16_t* Kb = (bf16_t*)p; p += matE * sizeof(bf16_t);
  bf16_t* Vb = (bf16_t*)p; p += matE * sizeof(bf16_t);
  bf16_t* Qr = (bf16_t*)p; p += matE * sizeof(bf16_t);
  bf16_t* Kr = (bf16_t*)p; p += matE * sizeof(bf16_t);
  bf16_t* Vt = (bf16_t*)p; p += matE * sizeof(bf16_t);
  bf16_t* Ob = Qb;  // Qb is dead after RoPE relayout; reuse for attention out

  auto blocksFor = [](size_t n) { return (unsigned)((n + 255) / 256); };

  // 1) fp32 -> bf16
  cvt_f32_bf16<<<blocksFor(matE), 256, 0, stream>>>(x, Xb, matE);
  cvt_f32_bf16<<<blocksFor(wE), 256, 0, stream>>>(Wq, Wb + 0 * wE, wE);
  cvt_f32_bf16<<<blocksFor(wE), 256, 0, stream>>>(Wk, Wb + 1 * wE, wE);
  cvt_f32_bf16<<<blocksFor(wE), 256, 0, stream>>>(Wv, Wb + 2 * wE, wE);
  cvt_f32_bf16<<<blocksFor(wE), 256, 0, stream>>>(Wo, Wb + 3 * wE, wE);

  // 2) Q/K/V projections (WMMA GEMM, bf16 out): 64x64 tiles, 8 waves/block
  const unsigned gemmBlocks = (unsigned)(((rows / 64) * (D / 64) + 7) / 8);
  gemm64_nt<true><<<gemmBlocks, 256, 0, stream>>>(Xb, Wb + 0 * wE, Qb,
                                                  (int)rows, D, D);
  gemm64_nt<true><<<gemmBlocks, 256, 0, stream>>>(Xb, Wb + 1 * wE, Kb,
                                                  (int)rows, D, D);
  gemm64_nt<true><<<gemmBlocks, 256, 0, stream>>>(Xb, Wb + 2 * wE, Vb,
                                                  (int)rows, D, D);

  // 3) RoPE + relayout
  const size_t pairs = (size_t)BATCH * S * N_HEADS * (D_K / 2);
  rope_relayout<<<blocksFor(pairs), 256, 0, stream>>>(Qb, Qr, tpos, S, 1);
  rope_relayout<<<blocksFor(pairs), 256, 0, stream>>>(Kb, Kr, tpos, S, 1);
  rope_relayout<<<blocksFor(pairs), 256, 0, stream>>>(Vb, Vt, tpos, S, 0);

  // 4) causal flash attention (one wave per (b,h,32-query block))
  dim3 ag(S / 32, N_HEADS, BATCH);
  flash_attn<<<ag, 32, 0, stream>>>(Qr, Kr, Vt, Ob, S);

  // 5) output projection -> fp32 d_out [B,S,D]
  gemm64_nt<false><<<gemmBlocks, 256, 0, stream>>>(Ob, Wb + 3 * wE, d_out,
                                                   (int)rows, D, D);
}
